// GraphAttention_76441827934506
// MI455X (gfx1250) — matrix operand order repaired and checked by reading, not compiled, not run on hardware
//
#include <hip/hip_runtime.h>
#include <hip/hip_bf16.h>

// Problem constants (match reference)
#define BB    4
#define NN    2048
#define IN_F  128
#define OUT_F 128
#define HH    4
#define DD    32
#define NEG_SLOPE 0.2f

typedef __attribute__((ext_vector_type(16))) _Float16 v16h;
typedef __attribute__((ext_vector_type(8)))  float    v8f;

union AV { v16h h; unsigned int u[8]; };

static __device__ __forceinline__ v8f wmma_f16(const AV& a, const AV& b, v8f c) {
  // 8 args: (neg_a, A, neg_b, B, c_mod, C, reuse_a, reuse_b)
  return __builtin_amdgcn_wmma_f32_16x16x32_f16(false, a.h, false, b.h,
                                                (short)0, c, false, false);
}

// ---------------------------------------------------------------------------
// Kernel 0a: f32 -> f16 conversion (x and W)
// ---------------------------------------------------------------------------
__global__ void cvt_f32_f16_kernel(const float* __restrict__ src,
                                   _Float16* __restrict__ dst, int n) {
  int i = blockIdx.x * 256 + threadIdx.x;
  if (i < n) dst[i] = (_Float16)src[i];
}

// ---------------------------------------------------------------------------
// Kernel 0b: bit-pack adjacency: adj_pk[n][w] bit j = (adj[n][w*32+j] != 0)
// ---------------------------------------------------------------------------
__global__ void pack_adj_kernel(const int* __restrict__ adj,
                                unsigned int* __restrict__ pk) {
  int t = blockIdx.x * 256 + threadIdx.x;       // t = n*(N/32) + w
  if (t >= NN * (NN / 32)) return;
  int n = t >> 6;                               // N/32 == 64
  int w = t & 63;
  const int* row = adj + (size_t)n * NN + w * 32;
  unsigned int bits = 0u;
  #pragma unroll 8
  for (int j = 0; j < 32; ++j) bits |= (row[j] != 0 ? 1u : 0u) << j;
  pk[t] = bits;
}

// ---------------------------------------------------------------------------
// Kernel 1: projection GEMM  h = x @ W^T  via v_wmma_f32_16x16x32_f16.
// Output stored transposed per head: h_t[b][head][d][n]  (f16), so that the
// attention kernel's B-matrix K-pairs (along m==n axis) are contiguous b32s.
// One wave per (16-row tile, 16-col tile); 4 waves per block.
// ---------------------------------------------------------------------------
__global__ __launch_bounds__(128)
void proj_kernel(const _Float16* __restrict__ xh,   // [B*N][IN_F]
                 const _Float16* __restrict__ wh,   // [OUT_F][IN_F]
                 _Float16* __restrict__ ht) {       // [B][H][D][N]
  const int lane = threadIdx.x & 31;
  const int wv   = threadIdx.x >> 5;
  const int wi   = blockIdx.x * 4 + wv;   // 0 .. 4095
  const int rt   = wi >> 3;               // row tile (0..511) over B*N
  const int ot   = wi & 7;                // output-col tile (0..7)
  const int hl   = lane >> 4;             // lane half
  const int lm   = lane & 15;

  const int arow = rt * 16 + lm;          // A row (flat b*N+n), lane-held
  const int bcol = ot * 16 + lm;          // B column (output feature)

  v8f acc = {};
  for (int k0 = 0; k0 < IN_F; k0 += 32) {
    AV a, b;
    #pragma unroll
    for (int v = 0; v < 8; ++v) {
      // A layout: VGPR v K-pair base = 2v + (v>=4 ? 8:0) + 8*hl
      int ka = k0 + 2 * v + ((v >= 4) ? 8 : 0) + 8 * hl;
      a.u[v] = *(const unsigned int*)(xh + (size_t)arow * IN_F + ka);
      // B layout: VGPR v K-pair base = 2v + (v>=4 ? 8:0) + 8*hl ; B[k][o] = W[o][k]
      int kb = k0 + 2 * v + ((v >= 4) ? 8 : 0) + 8 * hl;
      b.u[v] = *(const unsigned int*)(wh + (size_t)bcol * IN_F + kb);
    }
    acc = wmma_f16(a, b, acc);
  }

  // C layout: lane l, reg r -> row M = r + 8*hl, col = lm
  #pragma unroll
  for (int r = 0; r < 8; ++r) {
    int fr = rt * 16 + r + 8 * hl;        // flat row = b*N + n
    int b_ = fr >> 11;                    // / N
    int n_ = fr & (NN - 1);
    int oc = ot * 16 + lm;                // output feature
    int hd = oc >> 5;                     // head
    int d  = oc & (DD - 1);
    ht[(((size_t)b_ * HH + hd) * DD + d) * NN + n_] = (_Float16)acc[r];
  }
}

// ---------------------------------------------------------------------------
// Kernel 2: e_src / e_dst : per (b, head, n) dot over D of h_t with a-vectors
// Stored as [B*H][N] (contiguous in n -> coalesced, and row-loadable).
// ---------------------------------------------------------------------------
__global__ void compute_e_kernel(const _Float16* __restrict__ ht,
                                 const float* __restrict__ a_src,
                                 const float* __restrict__ a_dst,
                                 float* __restrict__ e_src,
                                 float* __restrict__ e_dst) {
  int t = blockIdx.x * 256 + threadIdx.x;   // (b*H + h)*N + n
  if (t >= BB * HH * NN) return;
  int n  = t & (NN - 1);
  int bh = t >> 11;
  int h  = bh & (HH - 1);
  float es = 0.f, ed = 0.f;
  #pragma unroll 8
  for (int d = 0; d < DD; ++d) {
    float hv = (float)ht[((size_t)bh * DD + d) * NN + n];
    es += hv * a_src[h * DD + d];
    ed += hv * a_dst[h * DD + d];
  }
  e_src[t] = es;
  e_dst[t] = ed;
}

// ---------------------------------------------------------------------------
// Kernel 3: fused masked-softmax attention + aggregation.
// Stability offset per row is the *bound* lrelu(e_src[n] + max_m e_dst[m])
// (leaky_relu is monotone, softmax is offset-invariant), computed during the
// cooperative e_dst load -> NO online rescaling: the inner loop has zero
// cross-lane ops and no accumulator scaling. P is produced directly in the
// WMMA-A f16 register layout; aggregation via v_wmma_f32_16x16x32_f16.
// ---------------------------------------------------------------------------
__global__ __launch_bounds__(128)
void attn_kernel(const _Float16* __restrict__ ht,         // [B][H][D][N]
                 const float* __restrict__ e_src,         // [B*H][N]
                 const float* __restrict__ e_dst,         // [B*H][N]
                 const unsigned int* __restrict__ adj_pk, // [N][N/32]
                 float* __restrict__ out) {               // [B][N][OUT_F]
  __shared__ float eds[NN];                               // 8 KB
  __shared__ float redbuf[128];

  const int tid  = threadIdx.x;
  const int lane = tid & 31;
  const int wv   = tid >> 5;
  const int bh   = blockIdx.y;                 // b*H + h
  const int b    = bh >> 2;
  const int h    = bh & (HH - 1);
  const int hl   = lane >> 4;
  const int lm   = lane & 15;

  // cooperative load of this (b,h)'s e_dst row + block max of e_dst
  float lmax = -__builtin_inff();
  for (int i = tid; i < NN; i += 128) {
    float v = e_dst[(size_t)bh * NN + i];
    eds[i] = v;
    lmax = fmaxf(lmax, v);
  }
  redbuf[tid] = lmax;
  __syncthreads();
  #pragma unroll
  for (int s = 64; s > 0; s >>= 1) {
    if (tid < s) redbuf[tid] = fmaxf(redbuf[tid], redbuf[tid + s]);
    __syncthreads();
  }
  const float gmax = redbuf[0];

  const int n0 = blockIdx.x * 64 + wv * 16;    // this wave's row tile base
  const int myrow = n0 + lm;                   // score row held by this lane
  const float es = e_src[(size_t)bh * NN + myrow];
  const unsigned int* adjrow = adj_pk + (size_t)myrow * (NN / 32);
  const _Float16* hb = ht + (size_t)bh * DD * NN;

  // fixed per-row stability offset: upper bound of the row's score max
  float mb = es + gmax;
  const float mrow = fmaxf(mb, NEG_SLOPE * mb);   // lrelu(es + gmax)

  float lrow = 0.f;                               // per-lane partial sum
  v8f acc0 = {}, acc1 = {};

  for (int m0 = 0; m0 < NN; m0 += 32) {
    // issue memory early: adj bits + both B tiles (h_t K-pairs contiguous)
    const unsigned int bits = adjrow[m0 >> 5];
    AV B0, B1;
    #pragma unroll
    for (int v = 0; v < 8; ++v) {
      int kb = m0 + 2 * v + ((v >= 4) ? 8 : 0) + 8 * hl;
      B0.u[v] = *(const unsigned int*)(hb + (size_t)(lm)      * NN + kb);
      B1.u[v] = *(const unsigned int*)(hb + (size_t)(16 + lm) * NN + kb);
    }

    // scores -> probabilities, produced directly in WMMA-A f16 layout
    AV A;
    #pragma unroll
    for (int v = 0; v < 8; ++v) {
      int Kb = 2 * v + ((v >= 4) ? 8 : 0) + 8 * hl;   // even -> float2 LDS load
      float2 ee = *(const float2*)&eds[m0 + Kb];
      float s0 = es + ee.x;  s0 = fmaxf(s0, NEG_SLOPE * s0);
      float s1 = es + ee.y;  s1 = fmaxf(s1, NEG_SLOPE * s1);
      float p0 = ((bits >> Kb)       & 1u) ? __expf(s0 - mrow) : 0.f;
      float p1 = ((bits >> (Kb + 1)) & 1u) ? __expf(s1 - mrow) : 0.f;
      lrow += p0 + p1;
      union { _Float16 f[2]; unsigned int u; } pk;
      pk.f[0] = (_Float16)p0;
      pk.f[1] = (_Float16)p1;
      A.u[v] = pk.u;
    }

    acc0 = wmma_f16(A, B0, acc0);
    acc1 = wmma_f16(A, B1, acc1);
  }

  // combine the two lane-halves of each row once, at the end
  lrow += __shfl_xor(lrow, 16, 32);
  const float inv = 1.0f / lrow;                 // diag guarantees lrow > 0

  // normalize and write out: row n = n0 + r + 8*hl, col d = {lm, 16+lm}
  #pragma unroll
  for (int r = 0; r < 8; ++r) {
    float iv = __shfl(inv, r + 8 * hl, 32);      // row stats live in lane (r+8*hl)
    int n = n0 + r + 8 * hl;
    float* orow = out + ((size_t)b * NN + n) * OUT_F + h * DD;
    orow[lm]      = acc0[r] * iv;
    orow[16 + lm] = acc1[r] * iv;
  }
}

// ---------------------------------------------------------------------------
// Launch
// ---------------------------------------------------------------------------
extern "C" void kernel_launch(void* const* d_in, const int* in_sizes, int n_in,
                              void* d_out, int out_size, void* d_ws, size_t ws_size,
                              hipStream_t stream) {
  const float* x     = (const float*)d_in[0];
  const int*   adj   = (const int*)d_in[1];
  const float* W     = (const float*)d_in[2];
  const float* a_src = (const float*)d_in[3];
  const float* a_dst = (const float*)d_in[4];
  float* out = (float*)d_out;

  // workspace carve-up (bytes)
  char* ws = (char*)d_ws;
  _Float16* xh     = (_Float16*)(ws);                               // 2 MB
  _Float16* wh     = (_Float16*)(ws + 2097152);                     // 32 KB
  _Float16* ht     = (_Float16*)(ws + 2097152 + 32768);             // 2 MB
  float*    e_src  = (float*)   (ws + 2097152 + 32768 + 2097152);   // 128 KB
  float*    e_dst  = (float*)   (ws + 2097152 + 32768 + 2097152 + 131072);
  unsigned int* adj_pk = (unsigned int*)(ws + 2097152 + 32768 + 2097152 + 2*131072);

  const int nx = BB * NN * IN_F;        // 1,048,576
  const int nw = OUT_F * IN_F;          // 16,384
  cvt_f32_f16_kernel<<<(nx + 255) / 256, 256, 0, stream>>>(x, xh, nx);
  cvt_f32_f16_kernel<<<(nw + 255) / 256, 256, 0, stream>>>(W, wh, nw);

  const int npk = NN * (NN / 32);       // 131,072
  pack_adj_kernel<<<(npk + 255) / 256, 256, 0, stream>>>(adj, adj_pk);

  // 4096 wave-tiles, 4 waves/block
  proj_kernel<<<1024, 128, 0, stream>>>(xh, wh, ht);

  const int ne = BB * HH * NN;          // 32,768
  compute_e_kernel<<<(ne + 255) / 256, 256, 0, stream>>>(ht, a_src, a_dst, e_src, e_dst);

  dim3 grid(NN / 64, BB * HH);          // (32, 16)
  attn_kernel<<<grid, 128, 0, stream>>>(ht, e_src, e_dst, adj_pk, out);
}
